// CNN_Mamba_6854767804862
// MI455X (gfx1250) — matrix operand order repaired and checked by reading, not compiled, not run on hardware
//
#include <hip/hip_runtime.h>
#include <hip/hip_bf16.h>

typedef _Float16 half_t;
typedef __attribute__((ext_vector_type(4)))  _Float16 v4h;
typedef __attribute__((ext_vector_type(8)))  _Float16 v8h;
typedef __attribute__((ext_vector_type(16))) _Float16 v16h;
typedef __attribute__((ext_vector_type(8)))  float    v8f;

static __device__ __forceinline__ int imin(int a, int b) { return a < b ? a : b; }
static __device__ __forceinline__ int imax(int a, int b) { return a > b ? a : b; }

// A-fragment (16x32 f16) per-lane LDS read:
//   row = lane&15 ; slots 0..15 -> K = base..base+15
//   where base = (lane&16) ? 16 : 0.   (contiguous K order, matches B fill)
static __device__ __forceinline__ v16h load_a_frag(const _Float16* Ash_row, int lane) {
  const int abase = (lane & 16) ? 16 : 0;
  v8h x0 = *(const v8h*)(Ash_row + abase);        // 16B aligned (row stride 80B)
  v8h x1 = *(const v8h*)(Ash_row + abase + 8);
  return __builtin_shufflevector(x0, x1, 0, 1, 2, 3, 4, 5, 6, 7,
                                 8, 9, 10, 11, 12, 13, 14, 15);
}

// ---------------------------------------------------------------------------
// Implicit-GEMM conv + BN + (residual) + (ReLU).  f16 WMMA, f32 accumulate.
// M = B*Ho*Wo, N = Cout, K = Cin*KH*KW.
// Block = 128 threads = 4 waves; all waves share one 16-row A tile (im2col)
// staged in LDS; wave w handles N-tile blockIdx.y*4 + w.
// ---------------------------------------------------------------------------
template <int KH, int KW>
__global__ void conv_bn_act_wmma(
    const float* __restrict__ in,  const float* __restrict__ wgt,
    const float* __restrict__ g,   const float* __restrict__ bb,
    const float* __restrict__ mr,  const float* __restrict__ vr,
    const float* __restrict__ res, float* __restrict__ out,
    int B, int Cin, int H, int W, int Cout,
    int stride, int pad, int Ho, int Wo, int relu)
{
  constexpr int KHW = KH * KW;
  __shared__ _Float16 Ash[16][40];            // 40-half stride: 16B-aligned, bank-spread

  const int tid  = threadIdx.x;
  const int lane = tid & 31;
  const int wave = tid >> 5;
  const int tm   = blockIdx.x;
  const int tn   = blockIdx.y * 4 + wave;
  const int M = B * Ho * Wo;
  const int K = Cin * KHW;

  // ---- loop-invariant staging coords: thread stages Ash[srow][skk..skk+3] ----
  const int srow = (tid * 4) >> 5;
  const int skk  = (tid * 4) & 31;
  const int sm   = imin(tm * 16 + srow, M - 1);
  const int sb   = sm / (Ho * Wo);
  const int srem = sm - sb * (Ho * Wo);
  const int soh  = srem / Wo;
  const int sow  = srem - soh * Wo;
  const int ih0  = soh * stride - pad;
  const int iw0  = sow * stride - pad;
  const float* inb = in + (long)sb * Cin * H * W;

  // ---- loop-invariant B coords ----
  const int n_b = tn * 16 + (lane & 15);
  const int n_c = imin(n_b, Cout - 1);
  const float* wrow = wgt + (long)n_c * K;
  const int kbase0 = (lane & 16) ? 16 : 0;

  v8f acc = {};
  for (int k0 = 0; k0 < K; k0 += 32) {
    // stage A tile (branchless clamped im2col gather)
    v4h h4;
#pragma unroll
    for (int j = 0; j < 4; ++j) {
      int k   = k0 + skk + j;
      int kcl = imin(k, K - 1);
      int cin = kcl / KHW;                 // compile-time divisor
      int rr  = kcl - cin * KHW;
      int kh  = rr / KW;
      int kw  = rr - kh * KW;
      int ih  = ih0 + kh, iw = iw0 + kw;
      bool vld = (k < K) & ((unsigned)ih < (unsigned)H) & ((unsigned)iw < (unsigned)W);
      int ihc = imin(imax(ih, 0), H - 1);
      int iwc = imin(imax(iw, 0), W - 1);
      float a = inb[(cin * H + ihc) * W + iwc];
      h4[j] = (half_t)(vld ? a : 0.f);
    }
    *(v4h*)&Ash[srow][skk] = h4;
    __syncthreads();

    v16h av = load_a_frag(&Ash[lane & 15][0], lane);

    v16h bv;
#pragma unroll
    for (int i = 0; i < 16; ++i) {
      int kb  = k0 + kbase0 + i;
      float w = wrow[imin(kb, K - 1)];
      bv[i] = (half_t)(kb < K ? w : 0.f);
    }
    __builtin_prefetch(wrow + k0 + 64, 0, 0);   // global_prefetch weight stream

    acc = __builtin_amdgcn_wmma_f32_16x16x32_f16(false, av, false, bv,
                                                 (short)0, acc, false, false);
    __syncthreads();
  }

#pragma unroll
  for (int r = 0; r < 8; ++r) {
    int m = tm * 16 + r + ((lane & 16) ? 8 : 0);
    int n = tn * 16 + (lane & 15);
    if (m < M && n < Cout) {
      float v = acc[r];
      v = (v - mr[n]) * rsqrtf(vr[n] + 1e-5f) * g[n] + bb[n];
      int b = m / (Ho * Wo); int rr2 = m - b * (Ho * Wo);
      long oi = (((long)b * Cout + n) * Ho + rr2 / Wo) * Wo + rr2 % Wo;
      if (res) v += res[oi];
      if (relu) v = v > 0.f ? v : 0.f;
      out[oi] = v;
    }
  }
}

// ---------------------------------------------------------------------------
// Generic GEMM: out[M,N](ldo) = A[M,K](lda) @ W[N,K]^T (+bias) (+softplus)
// Requires K % 32 == 0 and lda % 4 == 0 (true at every call site).
// Block = 128 threads = 4 waves sharing the A tile via LDS.
// ---------------------------------------------------------------------------
__global__ void gemm_wmma(const float* __restrict__ A, int lda,
                          const float* __restrict__ Wt,
                          const float* __restrict__ bias,
                          float* __restrict__ out, int ldo,
                          int M, int N, int K, int act)
{
  __shared__ _Float16 Ash[16][40];

  const int tid  = threadIdx.x;
  const int lane = tid & 31;
  const int wave = tid >> 5;
  const int tm   = blockIdx.x;
  const int tn   = blockIdx.y * 4 + wave;

  const int srow = (tid * 4) >> 5;
  const int skk  = (tid * 4) & 31;
  const int sm   = imin(tm * 16 + srow, M - 1);
  const float* sA = A + (long)sm * lda;

  const int n_b = tn * 16 + (lane & 15);
  const int n_c = imin(n_b, N - 1);
  const float* wrow = Wt + (long)n_c * K;
  const int kbase0 = (lane & 16) ? 16 : 0;

  v8f acc = {};
  for (int k0 = 0; k0 < K; k0 += 32) {
    float4 a4 = *(const float4*)(sA + k0 + skk);
    v4h h4 = { (half_t)a4.x, (half_t)a4.y, (half_t)a4.z, (half_t)a4.w };
    *(v4h*)&Ash[srow][skk] = h4;
    __syncthreads();

    v16h av = load_a_frag(&Ash[lane & 15][0], lane);

    const float4* wp = (const float4*)(wrow + k0 + kbase0);   // 64B aligned
    float4 w0 = wp[0], w1 = wp[1], w2 = wp[2], w3 = wp[3];
    v16h bv = { (half_t)w0.x, (half_t)w0.y, (half_t)w0.z, (half_t)w0.w,
                (half_t)w1.x, (half_t)w1.y, (half_t)w1.z, (half_t)w1.w,
                (half_t)w2.x, (half_t)w2.y, (half_t)w2.z, (half_t)w2.w,
                (half_t)w3.x, (half_t)w3.y, (half_t)w3.z, (half_t)w3.w };
    __builtin_prefetch(wrow + k0 + 64, 0, 0);

    acc = __builtin_amdgcn_wmma_f32_16x16x32_f16(false, av, false, bv,
                                                 (short)0, acc, false, false);
    __syncthreads();
  }

#pragma unroll
  for (int r = 0; r < 8; ++r) {
    int m = tm * 16 + r + ((lane & 16) ? 8 : 0);
    int n = tn * 16 + (lane & 15);
    if (m < M && n < N) {
      float v = acc[r];
      if (bias) v += bias[n];
      if (act == 1) v = (v > 20.f) ? v : log1pf(expf(v));   // softplus
      out[(long)m * ldo + n] = v;
    }
  }
}

// ---------------------------------------------------------------------------
// Max pool 3x3 stride 2 pad 1
// ---------------------------------------------------------------------------
__global__ void maxpool3x3s2(const float* __restrict__ in, float* __restrict__ out,
                             int B, int C, int H, int W, int Ho, int Wo)
{
  int idx = blockIdx.x * blockDim.x + threadIdx.x;
  int total = B * C * Ho * Wo;
  if (idx >= total) return;
  int wo = idx % Wo; int ho = (idx / Wo) % Ho; int c = (idx / (Wo * Ho)) % C;
  int b = idx / (Wo * Ho * C);
  float m = -3.402823466e38f;
  for (int kh = 0; kh < 3; ++kh)
    for (int kw = 0; kw < 3; ++kw) {
      int ih = ho * 2 - 1 + kh, iw = wo * 2 - 1 + kw;
      if (ih >= 0 && ih < H && iw >= 0 && iw < W) {
        float v = in[((b * C + c) * H + ih) * W + iw];
        m = v > m ? v : m;
      }
    }
  out[idx] = m;
}

// [B,C,P] -> [B,P,C]
__global__ void nchw_to_nlc(const float* __restrict__ x, float* __restrict__ o,
                            int B, int C, int P)
{
  int idx = blockIdx.x * blockDim.x + threadIdx.x;
  if (idx >= B * C * P) return;
  int p = idx % P; int c = (idx / P) % C; int b = idx / (P * C);
  o[((long)(b * P + p)) * C + c] = x[idx];
}

// causal depthwise conv1d (K=4) + bias + SiLU ; input is xi half of xz [B,L,2Di]
__global__ void dwconv_silu(const float* __restrict__ xz,
                            const float* __restrict__ cw,
                            const float* __restrict__ cb,
                            float* __restrict__ out, int B, int L, int Di)
{
  int idx = blockIdx.x * blockDim.x + threadIdx.x;
  if (idx >= B * L * Di) return;
  int c = idx % Di; int l = (idx / Di) % L; int b = idx / (Di * L);
  float s = cb[c];
#pragma unroll
  for (int j = 0; j < 4; ++j) {
    int ls = l - 3 + j;
    if (ls >= 0) s += xz[((long)(b * L + ls)) * (2 * Di) + c] * cw[c * 4 + j];
  }
  out[idx] = s / (1.f + expf(-s));
}

// selective-scan: one thread per (b, d_inner) channel; 16 states in registers
__global__ void ssm_scan(const float* __restrict__ dt,   // [B,L,Di] softplus'd
                         const float* __restrict__ Alog, // [Di,16]
                         const float* __restrict__ dbl,  // [B,L,64]: Bs@32, Cs@48
                         const float* __restrict__ xi,   // [B,L,Di]
                         const float* __restrict__ Dp,   // [Di]
                         const float* __restrict__ xz,   // [B,L,2Di], z at +Di
                         float* __restrict__ y,          // [B,L,Di]
                         int B, int L, int Di)
{
  int idx = blockIdx.x * blockDim.x + threadIdx.x;
  if (idx >= B * Di) return;
  int d = idx % Di; int b = idx / Di;
  float a[16], h[16];
#pragma unroll
  for (int s = 0; s < 16; ++s) { a[s] = -expf(Alog[d * 16 + s]); h[s] = 0.f; }
  float Dd = Dp[d];
  for (int l = 0; l < L; ++l) {
    long base = (long)(b * L + l);
    float dtv = dt[base * Di + d];
    float u   = xi[base * Di + d];
    const float* row = dbl + base * 64;
    float acc = 0.f;
#pragma unroll
    for (int s = 0; s < 16; ++s) {
      float dA = expf(dtv * a[s]);
      h[s] = dA * h[s] + dtv * row[32 + s] * u;
      acc += h[s] * row[48 + s];
    }
    float yy = acc + u * Dd;
    float z = xz[base * (2 * Di) + Di + d];
    yy *= z / (1.f + expf(-z));
    y[base * Di + d] = yy;
  }
}

// mean over L, LayerNorm, head. One block (512 thr) per batch.
__global__ void pool_ln_head(const float* __restrict__ x, // [B,L,512]
                             const float* __restrict__ lg, const float* __restrict__ lb,
                             const float* __restrict__ hw, const float* __restrict__ hb,
                             float* __restrict__ out, int B, int L)
{
  __shared__ float v[512];
  __shared__ float red[512];
  int b = blockIdx.x; int d = threadIdx.x;
  float s = 0.f;
  for (int l = 0; l < L; ++l) s += x[((long)(b * L + l)) * 512 + d];
  s /= (float)L;
  v[d] = s; red[d] = s;
  __syncthreads();
  for (int off = 256; off > 0; off >>= 1) { if (d < off) red[d] += red[d + off]; __syncthreads(); }
  float mu = red[0] / 512.f;
  __syncthreads();
  float dv = v[d] - mu;
  red[d] = dv * dv;
  __syncthreads();
  for (int off = 256; off > 0; off >>= 1) { if (d < off) red[d] += red[d + off]; __syncthreads(); }
  float var = red[0] / 512.f;
  __syncthreads();
  v[d] = dv * rsqrtf(var + 1e-5f) * lg[d] + lb[d];
  __syncthreads();
  if (d < 10) {
    float o = hb[d];
    for (int k = 0; k < 512; ++k) o += hw[d * 512 + k] * v[k];
    out[b * 10 + d] = o;
  }
}

// ---------------------------------------------------------------------------
static inline int cdiv(int a, int b) { return (a + b - 1) / b; }

extern "C" void kernel_launch(void* const* d_in, const int* in_sizes, int n_in,
                              void* d_out, int out_size, void* d_ws, size_t ws_size,
                              hipStream_t stream) {
  (void)in_sizes; (void)n_in; (void)out_size; (void)ws_size;
  const int B = 16, L = 49, DM = 512, DI = 1024, DS = 16, DTR = 32;

  // ---- unpack inputs in reference dict insertion order ----
  int cur = 0;
  auto nxt = [&]() { return (const float*)d_in[cur++]; };
  const float* x = nxt();
  const float* conv1 = nxt();
  const float* bn1[4]; for (int i = 0; i < 4; ++i) bn1[i] = nxt();
  struct Blk {
    const float *c1, *b1[4], *c2, *b2[4], *dc, *db[4]; bool down;
  } blk[4][2];
  for (int li = 0; li < 4; ++li)
    for (int bi = 0; bi < 2; ++bi) {
      Blk& k = blk[li][bi];
      k.c1 = nxt(); for (int i = 0; i < 4; ++i) k.b1[i] = nxt();
      k.c2 = nxt(); for (int i = 0; i < 4; ++i) k.b2[i] = nxt();
      k.down = (li > 0 && bi == 0);
      if (k.down) { k.dc = nxt(); for (int i = 0; i < 4; ++i) k.db[i] = nxt(); }
      else { k.dc = nullptr; for (int i = 0; i < 4; ++i) k.db[i] = nullptr; }
    }
  const float* project_w = nxt();
  const float* project_b = nxt();
  struct MP { const float *in_proj,*conv_w,*conv_b,*x_proj,*dt_proj,*dt_bias,*A_log,*D,*out_proj; } mp[2];
  for (int i = 0; i < 2; ++i) {
    mp[i].in_proj = nxt(); mp[i].conv_w = nxt(); mp[i].conv_b = nxt();
    mp[i].x_proj = nxt();  mp[i].dt_proj = nxt(); mp[i].dt_bias = nxt();
    mp[i].A_log = nxt();   mp[i].D = nxt();       mp[i].out_proj = nxt();
  }
  const float* ln_g = nxt(); const float* ln_b = nxt();
  const float* head_w = nxt(); const float* head_b = nxt();

  // ---- workspace bump allocator ----
  char* ws = (char*)d_ws; size_t off = 0;
  auto alloc = [&](size_t nfloat) {
    float* p = (float*)(ws + off);
    off += ((nfloat * sizeof(float) + 255) & ~(size_t)255);
    return p;
  };
  float* t_conv1 = alloc((size_t)B * 64 * 112 * 112);
  float* bufs[3]; for (int i = 0; i < 3; ++i) bufs[i] = alloc((size_t)B * 64 * 56 * 56);
  float* bufSkip = alloc((size_t)B * 128 * 28 * 28);
  float* xseq   = alloc((size_t)B * L * DM);
  float* seqA   = alloc((size_t)B * L * DM);
  float* seqB   = alloc((size_t)B * L * DM);
  float* xz     = alloc((size_t)B * L * 2 * DI);
  float* xiconv = alloc((size_t)B * L * DI);
  float* dbl    = alloc((size_t)B * L * 64);
  float* dtb    = alloc((size_t)B * L * DI);
  float* ybuf   = alloc((size_t)B * L * DI);

  // ---- stem: conv7x7 s2 + bn + relu ----
  {
    int M = B * 112 * 112;
    dim3 g(cdiv(M, 16), cdiv(64, 64));
    conv_bn_act_wmma<7, 7><<<g, 128, 0, stream>>>(x, conv1, bn1[0], bn1[1], bn1[2], bn1[3],
        nullptr, t_conv1, B, 3, 224, 224, 64, 2, 3, 112, 112, 1);
  }
  // maxpool 3x3 s2 pad1 -> bufs[0]
  {
    int total = B * 64 * 56 * 56;
    maxpool3x3s2<<<cdiv(total, 256), 256, 0, stream>>>(t_conv1, bufs[0], B, 64, 112, 112, 56, 56);
  }

  // ---- ResNet layers ----
  static const int CFG[4][3] = {{64, 64, 1}, {64, 128, 2}, {128, 256, 2}, {256, 512, 2}};
  int ci = 0; int curC = 64, curH = 56, curW = 56;
  for (int li = 0; li < 4; ++li) {
    int cout = CFG[li][1];
    for (int bi = 0; bi < 2; ++bi) {
      const Blk& k = blk[li][bi];
      int stride = (bi == 0) ? CFG[li][2] : 1;
      int Ho = (curH - 1) / stride + 1, Wo = (curW - 1) / stride + 1;
      float* cur_buf = bufs[ci];
      float* tmp1 = bufs[(ci + 1) % 3];
      float* tmp2 = bufs[(ci + 2) % 3];
      int M = B * Ho * Wo;
      dim3 g(cdiv(M, 16), cdiv(cout, 64));
      // conv1 3x3 + bn + relu
      conv_bn_act_wmma<3, 3><<<g, 128, 0, stream>>>(cur_buf, k.c1, k.b1[0], k.b1[1], k.b1[2], k.b1[3],
          nullptr, tmp1, B, curC, curH, curW, cout, stride, 1, Ho, Wo, 1);
      // shortcut
      const float* resp = cur_buf;
      if (k.down) {
        conv_bn_act_wmma<1, 1><<<g, 128, 0, stream>>>(cur_buf, k.dc, k.db[0], k.db[1], k.db[2], k.db[3],
            nullptr, bufSkip, B, curC, curH, curW, cout, stride, 0, Ho, Wo, 0);
        resp = bufSkip;
      }
      // conv2 3x3 + bn + add + relu
      conv_bn_act_wmma<3, 3><<<g, 128, 0, stream>>>(tmp1, k.c2, k.b2[0], k.b2[1], k.b2[2], k.b2[3],
          resp, tmp2, B, cout, Ho, Wo, cout, 1, 1, Ho, Wo, 1);
      ci = (ci + 2) % 3;
      curC = cout; curH = Ho; curW = Wo;
    }
  }

  // ---- [B,512,7,7] -> [B,49,512] ; project ----
  {
    int total = B * DM * L;
    nchw_to_nlc<<<cdiv(total, 256), 256, 0, stream>>>(bufs[ci], xseq, B, DM, L);
    dim3 g(cdiv(B * L, 16), cdiv(DM, 64));
    gemm_wmma<<<g, 128, 0, stream>>>(xseq, DM, project_w, project_b, seqA, DM,
                                     B * L, DM, DM, 0);
  }

  // ---- two Mamba blocks ----
  float* seq = seqA; float* seq_next = seqB;
  for (int mi = 0; mi < 2; ++mi) {
    const MP& m = mp[mi];
    int M = B * L;
    { dim3 g(cdiv(M, 16), cdiv(2 * DI, 64));
      gemm_wmma<<<g, 128, 0, stream>>>(seq, DM, m.in_proj, nullptr, xz, 2 * DI,
                                       M, 2 * DI, DM, 0); }
    { int total = M * DI;
      dwconv_silu<<<cdiv(total, 256), 256, 0, stream>>>(xz, m.conv_w, m.conv_b,
                                                        xiconv, B, L, DI); }
    { dim3 g(cdiv(M, 16), cdiv(64, 64));
      gemm_wmma<<<g, 128, 0, stream>>>(xiconv, DI, m.x_proj, nullptr, dbl, 64,
                                       M, DTR + 2 * DS, DI, 0); }
    { dim3 g(cdiv(M, 16), cdiv(DI, 64));
      gemm_wmma<<<g, 128, 0, stream>>>(dbl, 64, m.dt_proj, m.dt_bias, dtb, DI,
                                       M, DI, DTR, 1); }          // softplus
    { int total = B * DI;
      ssm_scan<<<cdiv(total, 256), 256, 0, stream>>>(dtb, m.A_log, dbl, xiconv,
                                                     m.D, xz, ybuf, B, L, DI); }
    { dim3 g(cdiv(M, 16), cdiv(DM, 64));
      gemm_wmma<<<g, 128, 0, stream>>>(ybuf, DI, m.out_proj, nullptr, seq_next, DM,
                                       M, DM, DI, 0); }
    float* t = seq; seq = seq_next; seq_next = t;
  }

  // ---- pool + LN + head ----
  pool_ln_head<<<B, 512, 0, stream>>>(seq, ln_g, ln_b, head_w, head_b,
                                      (float*)d_out, B, L);
}
